// CrossAttentionPoseRegression_12214886990553
// MI455X (gfx1250) — matrix operand order repaired and checked by reading, not compiled, not run on hardware
//
#include <hip/hip_runtime.h>

#define N_NODES 8192
#define N_EDGES 131072
#define HIDN 64
#define FEATD 144

typedef __attribute__((ext_vector_type(16))) _Float16 v16h;
typedef __attribute__((ext_vector_type(8)))  _Float16 h8;
typedef __attribute__((ext_vector_type(2)))  _Float16 h2v;
typedef __attribute__((ext_vector_type(8)))  float    v8f;

__device__ __forceinline__ v8f vzero8() {
  v8f z = {0.f,0.f,0.f,0.f,0.f,0.f,0.f,0.f};
  return z;
}
__device__ __forceinline__ h8 h8zero() {
  h8 z = {(_Float16)0.f,(_Float16)0.f,(_Float16)0.f,(_Float16)0.f,
          (_Float16)0.f,(_Float16)0.f,(_Float16)0.f,(_Float16)0.f};
  return z;
}

__device__ __forceinline__ v8f wmma16(v16h a, v16h b, v8f c) {
  return __builtin_amdgcn_wmma_f32_16x16x32_f16(false, a, false, b, (short)0, c, false, false);
}

// A tile 16x32 (f16) from row-major [m][k]; zeros for whole 8-chunks with k>=kmax
// (kmax is always a multiple of 8 here). Two ds_load_b128 per lane.
// Lane layout per CDNA5 ISA: lanes 0-15 row m, K {0..7,16..23}; lanes 16-31 row m, K {8..15,24..31}.
__device__ __forceinline__ v16h lda16(const _Float16* base, int ld, int koff, int kmax) {
  const int lane = threadIdx.x & 31;
  const int m  = lane & 15;
  const int kb = (lane & 16) ? 8 : 0;
  const int k0 = koff + kb;
  const int k1 = koff + 16 + kb;
  h8 lo = (k0 < kmax) ? *(const h8*)(base + m * ld + k0) : h8zero();
  h8 hi = (k1 < kmax) ? *(const h8*)(base + m * ld + k1) : h8zero();
  return __builtin_shufflevector(lo, hi, 0,1,2,3,4,5,6,7,8,9,10,11,12,13,14,15);
}

// B tile 32x16 from K-contiguous swizzled storage: Bsw[n*kstride + k] (zero-padded K).
// Lanes 0-15: col n, K {0..7,16..23} ; lanes 16-31: col n, K {8..15,24..31}.  Two ds_load_b128 per lane.
__device__ __forceinline__ v16h ldbsw(const _Float16* base, int kstride, int koff) {
  const int lane = threadIdx.x & 31;
  const int n  = lane & 15;
  const int kb = (lane & 16) ? 8 : 0;
  const _Float16* p = base + n * kstride + koff + kb;
  h8 lo = *(const h8*)p;
  h8 hi = *(const h8*)(p + 16);
  return __builtin_shufflevector(lo, hi, 0,1,2,3,4,5,6,7,8,9,10,11,12,13,14,15);
}

__device__ __forceinline__ float siluf(float x) { return x / (1.f + __expf(-x)); }

// ---------------------------------------------------------------- emb_in
// h(N,32) @ W(32,64) + b -> h64.  128 thr = 4 waves, 64 rows/block.
__global__ __launch_bounds__(128) void k_emb_in(const float* __restrict__ hin,
                                                const float* __restrict__ W,
                                                const float* __restrict__ B,
                                                float* __restrict__ hout) {
  __shared__ _Float16 sW[64 * 32];   // swizzled [n][k]
  __shared__ float sB[64];
  __shared__ _Float16 sA[4][16 * 32];
  const int tid = threadIdx.x, lane = tid & 31, wid = tid >> 5;
  for (int i = tid; i < 64 * 32; i += 128) {
    int n = i >> 5, k = i & 31;
    sW[i] = (_Float16)W[k * 64 + n];
  }
  if (tid < 64) sB[tid] = B[tid];
  const int rb = blockIdx.x * 64 + wid * 16;
  for (int i = lane; i < 16 * 16; i += 32) {
    int r = i >> 4, cp = (i & 15) * 2;
    float2 v = *(const float2*)(hin + (rb + r) * 32 + cp);
    h2v p = { (_Float16)v.x, (_Float16)v.y };
    *(h2v*)(&sA[wid][r * 32 + cp]) = p;
  }
  __syncthreads();
  v16h a = lda16(sA[wid], 32, 0, 32);
  const int mb = (lane & 16) ? 8 : 0, nn = lane & 15;
#pragma unroll
  for (int nt = 0; nt < 4; ++nt) {
    v16h b = ldbsw(sW + nt * 16 * 32, 32, 0);
    v8f acc = wmma16(a, b, vzero8());
#pragma unroll
    for (int v = 0; v < 8; ++v)
      hout[(rb + mb + v) * 64 + nt * 16 + nn] = acc[v] + sB[nt * 16 + nn];
  }
}

// ---------------------------------------------------------------- emb_out + concat coords
// hs(N,35): cols 0..31 = h @ W(64,32)+b ; cols 32..34 = x
__global__ __launch_bounds__(128) void k_emb_out(const float* __restrict__ h,
                                                 const float* __restrict__ W,
                                                 const float* __restrict__ B,
                                                 const float* __restrict__ x,
                                                 float* __restrict__ hs) {
  __shared__ _Float16 sW[32 * 64];   // swizzled [n][k]
  __shared__ float sB[32];
  __shared__ _Float16 sA[4][16 * 64];
  const int tid = threadIdx.x, lane = tid & 31, wid = tid >> 5;
  for (int i = tid; i < 32 * 64; i += 128) {
    int n = i >> 6, k = i & 63;
    sW[i] = (_Float16)W[k * 32 + n];
  }
  if (tid < 32) sB[tid] = B[tid];
  const int rb = blockIdx.x * 64 + wid * 16;
  for (int i = lane; i < 16 * 32; i += 32) {
    int r = i >> 5, cp = (i & 31) * 2;
    float2 v = *(const float2*)(h + (rb + r) * 64 + cp);
    h2v p = { (_Float16)v.x, (_Float16)v.y };
    *(h2v*)(&sA[wid][r * 64 + cp]) = p;
  }
  __syncthreads();
  const int mb = (lane & 16) ? 8 : 0, nn = lane & 15;
  v8f acc[2];
  acc[0] = vzero8(); acc[1] = vzero8();
#pragma unroll
  for (int kk = 0; kk < 2; ++kk) {
    v16h a = lda16(sA[wid], 64, kk * 32, 64);
#pragma unroll
    for (int nt = 0; nt < 2; ++nt) {
      v16h b = ldbsw(sW + nt * 16 * 64, 64, kk * 32);
      acc[nt] = wmma16(a, b, acc[nt]);
    }
  }
#pragma unroll
  for (int nt = 0; nt < 2; ++nt)
#pragma unroll
    for (int v = 0; v < 8; ++v)
      hs[(rb + mb + v) * 35 + nt * 16 + nn] = acc[nt][v] + sB[nt * 16 + nn];
  for (int i = tid; i < 64 * 3; i += 128) {
    int r = blockIdx.x * 64 + i / 3, d = i % 3;
    hs[r * 35 + 32 + d] = x[r * 3 + d];
  }
}

// ---------------------------------------------------------------- fused edge kernel
struct EdgeArgs {
  const int* edges;
  const float* eattr;
  const float* h;
  const float* x;
  float* agg;
  float* cacc;
  const float* ew1; const float* eb1;
  const float* ew2; const float* eb2;
  const float* lng; const float* lnb;
  const float* cw1; const float* cb1; const float* cw2;
};

__global__ __launch_bounds__(128) void k_edge(EdgeArgs A_) {
  __shared__ _Float16 sW1[64 * 160];     // edge_w1 heads concat, swizzled [n][k], K padded 144->160
  __shared__ _Float16 sW2[4 * 16 * 32];  // edge_w2 per head, swizzled [o][d], K padded 16->32
  __shared__ _Float16 sCW1[64 * 64];     // coord_w1 swizzled [n][k]
  __shared__ float sB1[64], sB2[64], sLG[64], sLB[64], sCB1[64], sCW2[64];
  __shared__ _Float16 sF[4][16 * FEATD]; // per-wave feature/hdn/LN tile (row-major)
  __shared__ float sCD[4][48];           // coord_diff per edge
  __shared__ float sPhi[4][16];
  __shared__ int sRow[4][16], sCol[4][16];

  const int tid = threadIdx.x, lane = tid & 31, wid = tid >> 5;

  for (int i = tid; i < 64 * 160; i += 128) {
    int n = i / 160, k = i - n * 160;
    int hh = n >> 4, d = n & 15;
    sW1[i] = (k < FEATD) ? (_Float16)A_.ew1[(hh * FEATD + k) * 16 + d] : (_Float16)0.f;
  }
  for (int i = tid; i < 2048; i += 128) {
    int hh = i >> 9, rem = i & 511;
    int n = rem >> 5, k = rem & 31;
    sW2[i] = (k < 16) ? (_Float16)A_.ew2[hh * 256 + k * 16 + n] : (_Float16)0.f;
  }
  for (int i = tid; i < 4096; i += 128) {
    int n = i >> 6, k = i & 63;
    sCW1[i] = (_Float16)A_.cw1[k * 64 + n];
  }
  if (tid < 64) {
    sB1[tid] = A_.eb1[tid]; sB2[tid] = A_.eb2[tid];
    sLG[tid] = A_.lng[tid]; sLB[tid] = A_.lnb[tid];
    sCB1[tid] = A_.cb1[tid]; sCW2[tid] = A_.cw2[tid];
  }

  const int eb = blockIdx.x * 64 + wid * 16;

  if (lane < 16) {
    int e = eb + lane;
    int r = A_.edges[e];
    int c = A_.edges[N_EDGES + e];
    sRow[wid][lane] = r; sCol[wid][lane] = c;
    float xi0 = A_.x[r*3+0], xi1 = A_.x[r*3+1], xi2 = A_.x[r*3+2];
    float xk0 = A_.x[c*3+0], xk1 = A_.x[c*3+1], xk2 = A_.x[c*3+2];
    float r0 = xi0-xk0, r1 = xi1-xk1, r2 = xi2-xk2;
    float radial = r0*r0 + r1*r1 + r2*r2;
    float dist = sqrtf(radial);
    float dt = xi0*xk0 + xi1*xk1 + xi2*xk2;
    float ia = 1.f / (dist + 1e-8f);
    float a0 = r0*ia, a1 = r1*ia, a2 = r2*ia;
    float cp0 = xi1*xk2 - xi2*xk1;
    float cp1 = xi2*xk0 - xi0*xk2;
    float cp2 = xi0*xk1 - xi1*xk0;
    float cpn = sqrtf(cp0*cp0 + cp1*cp1 + cp2*cp2);
    float ib = 1.f / (cpn + 1e-8f);
    float b0 = cp0*ib, b1 = cp1*ib, b2 = cp2*ib;
    float c0 = a1*b2 - a2*b1, c1 = a2*b0 - a0*b2, c2 = a0*b1 - a1*b0;
    float na = sqrtf(a0*a0 + a1*a1 + a2*a2);
    float nb = sqrtf(b0*b0 + b1*b1 + b2*b2);
    float nc = sqrtf(c0*c0 + c1*c1 + c2*c2);
    bool mask = (na < 1e-6f) || (nb < 1e-6f) || (nc < 1e-6f);
    float M[9];
    if (mask) {
      M[0]=1.f;M[1]=0.f;M[2]=0.f;M[3]=0.f;M[4]=1.f;M[5]=0.f;M[6]=0.f;M[7]=0.f;M[8]=1.f;
    } else {
      M[0]=a0;M[1]=b0;M[2]=c0;M[3]=a1;M[4]=b1;M[5]=c1;M[6]=a2;M[7]=b2;M[8]=c2;
    }
    _Float16* fr = &sF[wid][lane * FEATD];
    fr[128] = (_Float16)radial;
    fr[129] = (_Float16)dist;
    fr[130] = (_Float16)dt;
#pragma unroll
    for (int j = 0; j < 9; ++j) fr[131 + j] = (_Float16)M[j];
#pragma unroll
    for (int j = 0; j < 4; ++j) fr[140 + j] = (_Float16)A_.eattr[e * 4 + j];
    sCD[wid][lane*3+0] = r0; sCD[wid][lane*3+1] = r1; sCD[wid][lane*3+2] = r2;
  }
  __syncthreads();

  // gather h[row], h[col] (float2 global loads, packed f16 LDS stores)
  for (int i = lane; i < 16 * 32; i += 32) {
    int e = i >> 5, cp = (i & 31) * 2;
    float2 v1 = *(const float2*)(A_.h + sRow[wid][e] * 64 + cp);
    float2 v2 = *(const float2*)(A_.h + sCol[wid][e] * 64 + cp);
    h2v p1 = { (_Float16)v1.x, (_Float16)v1.y };
    h2v p2 = { (_Float16)v2.x, (_Float16)v2.y };
    *(h2v*)(&sF[wid][e * FEATD + cp]) = p1;
    *(h2v*)(&sF[wid][e * FEATD + 64 + cp]) = p2;
  }
  __syncthreads();

  const int mb = (lane & 16) ? 8 : 0, nn = lane & 15;

  // GEMM1: feat(16x144) @ W1(144x64)
  v8f acc[4];
#pragma unroll
  for (int nt = 0; nt < 4; ++nt) acc[nt] = vzero8();
#pragma unroll
  for (int kk = 0; kk < 5; ++kk) {
    v16h a = lda16(sF[wid], FEATD, kk * 32, FEATD);
#pragma unroll
    for (int nt = 0; nt < 4; ++nt) {
      v16h b = ldbsw(sW1 + nt * 16 * 160, 160, kk * 32);
      acc[nt] = wmma16(a, b, acc[nt]);
    }
  }
  __syncthreads();
  // silu(+b1) -> hdn f16 into cols 0..63 of sF
#pragma unroll
  for (int nt = 0; nt < 4; ++nt)
#pragma unroll
    for (int v = 0; v < 8; ++v) {
      int col = nt * 16 + nn;
      sF[wid][(mb + v) * FEATD + col] = (_Float16)siluf(acc[nt][v] + sB1[col]);
    }
  __syncthreads();

  // GEMM2: per-head 16x16 (K zero-padded to 32)
  v8f a2[4];
#pragma unroll
  for (int hh = 0; hh < 4; ++hh) {
    v16h a = lda16(sF[wid] + hh * 16, FEATD, 0, 16);
    v16h b = ldbsw(sW2 + hh * 512, 32, 0);
    a2[hh] = wmma16(a, b, vzero8());
  }

  // +b2, LayerNorm over 64 cols in registers; scatter agg; stash f16 LN for coord GEMM
  float val2[4][8];
  float rs[8], rq[8];
#pragma unroll
  for (int v = 0; v < 8; ++v) { rs[v] = 0.f; rq[v] = 0.f; }
#pragma unroll
  for (int hh = 0; hh < 4; ++hh)
#pragma unroll
    for (int v = 0; v < 8; ++v) {
      float val = a2[hh][v] + sB2[hh * 16 + nn];
      val2[hh][v] = val; rs[v] += val; rq[v] += val * val;
    }
#pragma unroll
  for (int v = 0; v < 8; ++v)
#pragma unroll
    for (int off = 8; off >= 1; off >>= 1) {
      rs[v] += __shfl_xor(rs[v], off, 16);
      rq[v] += __shfl_xor(rq[v], off, 16);
    }
  __syncthreads();
#pragma unroll
  for (int v = 0; v < 8; ++v) {
    float mean = rs[v] * (1.f / 64.f);
    float var = rq[v] * (1.f / 64.f) - mean * mean;
    float rstd = rsqrtf(var + 1e-5f);
    int row = mb + v;
#pragma unroll
    for (int hh = 0; hh < 4; ++hh) {
      int col = hh * 16 + nn;
      float nval = (val2[hh][v] - mean) * rstd * sLG[col] + sLB[col];
      atomicAdd(&A_.agg[sRow[wid][row] * 64 + col], nval);
      sF[wid][row * FEATD + col] = (_Float16)nval;
    }
  }
  __syncthreads();

  // coord MLP: silu(ef @ cw1 + cb1) @ cw2 -> phi
  v8f a3[4];
#pragma unroll
  for (int nt = 0; nt < 4; ++nt) a3[nt] = vzero8();
#pragma unroll
  for (int kk = 0; kk < 2; ++kk) {
    v16h a = lda16(sF[wid], FEATD, kk * 32, 64);
#pragma unroll
    for (int nt = 0; nt < 4; ++nt) {
      v16h b = ldbsw(sCW1 + nt * 16 * 64, 64, kk * 32);
      a3[nt] = wmma16(a, b, a3[nt]);
    }
  }
  float ps[8];
#pragma unroll
  for (int v = 0; v < 8; ++v) ps[v] = 0.f;
#pragma unroll
  for (int nt = 0; nt < 4; ++nt) {
    int col = nt * 16 + nn;
    float w = sCW2[col], bb = sCB1[col];
#pragma unroll
    for (int v = 0; v < 8; ++v) ps[v] += siluf(a3[nt][v] + bb) * w;
  }
#pragma unroll
  for (int v = 0; v < 8; ++v)
#pragma unroll
    for (int off = 8; off >= 1; off >>= 1) ps[v] += __shfl_xor(ps[v], off, 16);
  if (nn == 0) {
#pragma unroll
    for (int v = 0; v < 8; ++v) sPhi[wid][mb + v] = ps[v];
  }
  __syncthreads();
  for (int i = lane; i < 48; i += 32) {
    int e = i / 3, d = i - e * 3;
    atomicAdd(&A_.cacc[sRow[wid][e] * 3 + d], sCD[wid][i] * sPhi[wid][e]);
  }
}

// ---------------------------------------------------------------- node update
__global__ __launch_bounds__(128) void k_node(const float* __restrict__ h,
                                              const float* __restrict__ agg,
                                              const float* __restrict__ cacc,
                                              const float* __restrict__ W1,
                                              const float* __restrict__ B1,
                                              const float* __restrict__ W2,
                                              const float* __restrict__ B2,
                                              float* __restrict__ hout,
                                              float* __restrict__ x) {
  __shared__ _Float16 sW1[64 * 128];  // swizzled [n][k]
  __shared__ _Float16 sW2[64 * 64];   // swizzled [n][k]
  __shared__ float sB1[64], sB2[64];
  __shared__ _Float16 sM[4][16 * 128];
  __shared__ _Float16 sT[4][16 * 64];
  const int tid = threadIdx.x, lane = tid & 31, wid = tid >> 5;
  for (int i = tid; i < 64 * 128; i += 128) {
    int n = i >> 7, k = i & 127;
    sW1[i] = (_Float16)W1[k * 64 + n];
  }
  for (int i = tid; i < 64 * 64; i += 128) {
    int n = i >> 6, k = i & 63;
    sW2[i] = (_Float16)W2[k * 64 + n];
  }
  if (tid < 64) { sB1[tid] = B1[tid]; sB2[tid] = B2[tid]; }
  const int nb = blockIdx.x * 64 + wid * 16;
  for (int i = lane; i < 16 * 64; i += 32) {
    int r = i >> 6, c = (i & 63) * 2;
    float2 v = (c < 64) ? *(const float2*)(h + (nb + r) * 64 + c)
                        : *(const float2*)(agg + (nb + r) * 64 + (c - 64));
    h2v p = { (_Float16)v.x, (_Float16)v.y };
    *(h2v*)(&sM[wid][r * 128 + c]) = p;
  }
  __syncthreads();
  const int mb = (lane & 16) ? 8 : 0, nn = lane & 15;
  v8f acc[4];
#pragma unroll
  for (int nt = 0; nt < 4; ++nt) acc[nt] = vzero8();
#pragma unroll
  for (int kk = 0; kk < 4; ++kk) {
    v16h a = lda16(sM[wid], 128, kk * 32, 128);
#pragma unroll
    for (int nt = 0; nt < 4; ++nt) {
      v16h b = ldbsw(sW1 + nt * 16 * 128, 128, kk * 32);
      acc[nt] = wmma16(a, b, acc[nt]);
    }
  }
#pragma unroll
  for (int nt = 0; nt < 4; ++nt)
#pragma unroll
    for (int v = 0; v < 8; ++v) {
      int col = nt * 16 + nn;
      sT[wid][(mb + v) * 64 + col] = (_Float16)siluf(acc[nt][v] + sB1[col]);
    }
  __syncthreads();
  v8f o[4];
#pragma unroll
  for (int nt = 0; nt < 4; ++nt) o[nt] = vzero8();
#pragma unroll
  for (int kk = 0; kk < 2; ++kk) {
    v16h a = lda16(sT[wid], 64, kk * 32, 64);
#pragma unroll
    for (int nt = 0; nt < 4; ++nt) {
      v16h b = ldbsw(sW2 + nt * 16 * 64, 64, kk * 32);
      o[nt] = wmma16(a, b, o[nt]);
    }
  }
#pragma unroll
  for (int nt = 0; nt < 4; ++nt)
#pragma unroll
    for (int v = 0; v < 8; ++v) {
      int row = nb + mb + v, col = nt * 16 + nn;
      hout[row * 64 + col] = h[row * 64 + col] + o[nt][v] + sB2[col];
    }
  for (int i = tid; i < 64 * 3; i += 128) {
    int r = blockIdx.x * 64 + i / 3, d = i % 3;
    x[r * 3 + d] += cacc[r * 3 + d];
  }
}

// ---------------------------------------------------------------- corr loss
__global__ void k_corr(const float* __restrict__ hs, const float* __restrict__ ht,
                       const int* __restrict__ corr, const float* __restrict__ lab,
                       float* accp) {
  int k = blockIdx.x * 256 + threadIdx.x;
  if (k < 1024) {
    int i = corr[2 * k], j = corr[2 * k + 1];
    float num = 0.f, na = 0.f, nb = 0.f;
    for (int m = 0; m < 35; ++m) {
      float a = hs[i * 35 + m], b = ht[j * 35 + m];
      num += a * b; na += a * a; nb += b * b;
    }
    float sim = num / (fmaxf(sqrtf(na), 1e-12f) * fmaxf(sqrtf(nb), 1e-12f));
    float d = sim - lab[k];
    atomicAdd(accp, d * d);
  }
}

// ---------------------------------------------------------------- hs^T (f16, 48x8192, zero-padded rows)
__global__ void k_transpose(const float* __restrict__ hs, _Float16* __restrict__ At) {
  int idx = blockIdx.x * 256 + threadIdx.x;
  if (idx < 48 * 8192) {
    int m = idx >> 13, kk = idx & 8191;
    At[idx] = (m < 35) ? (_Float16)hs[kk * 35 + m] : (_Float16)0.f;
  }
}

// ---------------------------------------------------------------- compress GEMM1: (48x8192)@(8192x2048), relu(+b1)
__global__ __launch_bounds__(128) void k_comp1(const _Float16* __restrict__ At,
                                               const float* __restrict__ W,
                                               const float* __restrict__ B1,
                                               float* __restrict__ C) {
  __shared__ _Float16 sA[16 * 32];
  __shared__ _Float16 sB[64 * 32];  // swizzled [n][k]
  const int tid = threadIdx.x, lane = tid & 31, wid = tid >> 5;
  const int mt = blockIdx.y;
  const int nb = blockIdx.x * 64;
  v8f acc = vzero8();
  for (int kk = 0; kk < 256; ++kk) {
    int k0 = kk * 32;
    if (tid < 32 && k0 + 32 < 8192)
      __builtin_prefetch((const void*)(W + (size_t)(k0 + 32 + tid) * 2048 + nb), 0, 1);
    for (int i = tid; i < 64; i += 128) {
      int r = i >> 2, c8 = (i & 3) * 8;
      *(h8*)(&sA[r * 32 + c8]) = *(const h8*)(At + (mt * 16 + r) * 8192 + k0 + c8);
    }
    for (int i = tid; i < 2048; i += 128) {
      int r = i >> 6, c = i & 63;
      sB[c * 32 + r] = (_Float16)W[(k0 + r) * 2048 + nb + c];
    }
    __syncthreads();
    v16h a = lda16(sA, 32, 0, 32);
    v16h b = ldbsw(sB + wid * 16 * 32, 32, 0);
    acc = wmma16(a, b, acc);
    __syncthreads();
  }
  const int mb = (lane & 16) ? 8 : 0, nn = lane & 15;
#pragma unroll
  for (int v = 0; v < 8; ++v) {
    int m = mt * 16 + mb + v, n = nb + wid * 16 + nn;
    float val = (m < 35) ? fmaxf(acc[v] + B1[n], 0.f) : 0.f;
    C[m * 2048 + n] = val;
  }
}

// ---------------------------------------------------------------- compress GEMM2: (48x2048)@(2048x128)+b2
__global__ __launch_bounds__(128) void k_comp2(const float* __restrict__ T,
                                               const float* __restrict__ W,
                                               const float* __restrict__ B2,
                                               float* __restrict__ O) {
  __shared__ _Float16 sA[16 * 32];
  __shared__ _Float16 sB[64 * 32];  // swizzled [n][k]
  const int tid = threadIdx.x, lane = tid & 31, wid = tid >> 5;
  const int mt = blockIdx.y;
  const int nb = blockIdx.x * 64;
  v8f acc = vzero8();
  for (int kk = 0; kk < 64; ++kk) {
    int k0 = kk * 32;
    for (int i = tid; i < 256; i += 128) {
      int r = i >> 4, cp = (i & 15) * 2;
      float2 v = *(const float2*)(T + (mt * 16 + r) * 2048 + k0 + cp);
      h2v p = { (_Float16)v.x, (_Float16)v.y };
      *(h2v*)(&sA[r * 32 + cp]) = p;
    }
    for (int i = tid; i < 2048; i += 128) {
      int r = i >> 6, c = i & 63;
      sB[c * 32 + r] = (_Float16)W[(k0 + r) * 128 + nb + c];
    }
    __syncthreads();
    v16h a = lda16(sA, 32, 0, 32);
    v16h b = ldbsw(sB + wid * 16 * 32, 32, 0);
    acc = wmma16(a, b, acc);
    __syncthreads();
  }
  const int mb = (lane & 16) ? 8 : 0, nn = lane & 15;
#pragma unroll
  for (int v = 0; v < 8; ++v) {
    int m = mt * 16 + mb + v, n = nb + wid * 16 + nn;
    O[m * 128 + n] = (m < 35) ? (acc[v] + B2[n]) : 0.f;
  }
}

// ---------------------------------------------------------------- column l2norm: cs[j][m] = O[m][j]/||O[:,j]||
__global__ void k_norm(const float* __restrict__ O, float* __restrict__ cs) {
  int j = threadIdx.x;  // 128
  float s = 0.f;
  for (int m = 0; m < 35; ++m) { float v = O[m * 128 + j]; s += v * v; }
  float inv = 1.f / fmaxf(sqrtf(s), 1e-12f);
  for (int m = 0; m < 35; ++m) cs[j * 48 + m] = O[m * 128 + j] * inv;
}

// ---------------------------------------------------------------- sim = cs @ ct^T (128x128)
__global__ void k_sim(const float* __restrict__ cs, const float* __restrict__ ct,
                      float* __restrict__ S) {
  int idx = blockIdx.x * 256 + threadIdx.x;
  int p = idx >> 7, q = idx & 127;
  float s = 0.f;
  for (int m = 0; m < 35; ++m) s += cs[p * 48 + m] * ct[q * 48 + m];
  S[idx] = s;
}

// ---------------------------------------------------------------- one-sided Jacobi SVD (128x128) + final loss
__global__ __launch_bounds__(256) void k_jacobi(const float* __restrict__ S,
                                                const float* __restrict__ corr_acc,
                                                float* __restrict__ out) {
  __shared__ float A[128 * 128];
  const int tid = threadIdx.x;
  for (int i = tid; i < 16384; i += 256) A[i] = S[i];
  __syncthreads();
  const int k = tid >> 2, t4 = tid & 3;
  for (int sweep = 0; sweep < 10; ++sweep) {
    for (int r = 0; r < 127; ++r) {
      int a_ = (r + k) % 127;
      int b_ = (k == 0) ? 127 : (r - k + 127) % 127;
      int p = a_ < b_ ? a_ : b_;
      int q = a_ < b_ ? b_ : a_;
      float app = 0.f, aqq = 0.f, apq = 0.f;
      for (int i = t4; i < 128; i += 4) {
        float xx = A[i * 128 + p], yy = A[i * 128 + q];
        app += xx * xx; aqq += yy * yy; apq += xx * yy;
      }
      app += __shfl_xor(app, 1); app += __shfl_xor(app, 2);
      aqq += __shfl_xor(aqq, 1); aqq += __shfl_xor(aqq, 2);
      apq += __shfl_xor(apq, 1); apq += __shfl_xor(apq, 2);
      float c = 1.f, s = 0.f;
      if (fabsf(apq) > 1e-20f) {
        float z = (aqq - app) / (2.f * apq);
        float t = ((z >= 0.f) ? 1.f : -1.f) / (fabsf(z) + sqrtf(1.f + z * z));
        c = rsqrtf(1.f + t * t); s = c * t;
      }
      for (int i = t4; i < 128; i += 4) {
        float xx = A[i * 128 + p], yy = A[i * 128 + q];
        A[i * 128 + p] = c * xx - s * yy;
        A[i * 128 + q] = s * xx + c * yy;
      }
      __syncthreads();
    }
  }
  float nsq = 0.f;
  if (tid < 128)
    for (int i = 0; i < 128; ++i) { float xx = A[i * 128 + tid]; nsq += xx * xx; }
  __syncthreads();
  if (tid < 128) { float d = sqrtf(nsq) - 1.f; A[tid] = d * d; }
  __syncthreads();
  if (tid == 0) {
    float s = 0.f;
    for (int j = 0; j < 128; ++j) s += A[j];
    out[0] = corr_acc[0] * (1.f / 1024.f) + s * (1.f / 128.f);
  }
}

// ================================================================ host
extern "C" void kernel_launch(void* const* d_in, const int* in_sizes, int n_in,
                              void* d_out, int out_size, void* d_ws, size_t ws_size,
                              hipStream_t stream) {
  (void)in_sizes; (void)n_in; (void)out_size; (void)ws_size;
  const float* h_src = (const float*)d_in[0];
  const float* x_src = (const float*)d_in[1];
  const int*   e_src = (const int*)d_in[2];
  const float* a_src = (const float*)d_in[3];
  const float* h_tgt = (const float*)d_in[4];
  const float* x_tgt = (const float*)d_in[5];
  const int*   e_tgt = (const int*)d_in[6];
  const float* a_tgt = (const float*)d_in[7];
  const int*   corr  = (const int*)d_in[8];
  const float* labels = (const float*)d_in[9];
  const float* pp = (const float*)d_in[10];

  // params flattened in JAX tree order (sorted dict keys, lists in order)
  const float* comp_b1 = pp; pp += 2048;
  const float* comp_b2 = pp; pp += 128;
  const float* comp_w1 = pp; pp += 8192 * 2048;
  const float* comp_w2 = pp; pp += 2048 * 128;
  const float* emb_in_b = pp; pp += 64;
  const float* emb_in_w = pp; pp += 32 * 64;
  const float* emb_out_b = pp; pp += 32;
  const float* emb_out_w = pp; pp += 64 * 32;
  struct Lp { const float *cb1,*cw1,*cw2,*eb1,*eb2,*ew1,*ew2,*lnb,*lng,*nb1,*nb2,*nw1,*nw2; } L[5];
  for (int l = 0; l < 5; ++l) {
    L[l].cb1 = pp; pp += 64;
    L[l].cw1 = pp; pp += 4096;
    L[l].cw2 = pp; pp += 64;
    L[l].eb1 = pp; pp += 64;
    L[l].eb2 = pp; pp += 64;
    L[l].ew1 = pp; pp += 9216;
    L[l].ew2 = pp; pp += 1024;
    L[l].lnb = pp; pp += 64;
    L[l].lng = pp; pp += 64;
    L[l].nb1 = pp; pp += 64;
    L[l].nb2 = pp; pp += 64;
    L[l].nw1 = pp; pp += 8192;
    L[l].nw2 = pp; pp += 4096;
  }

  char* wsp = (char*)d_ws;
  auto carve = [&](size_t bytes) -> void* {
    void* r = (void*)wsp;
    wsp += (bytes + 255) & ~(size_t)255;
    return r;
  };
  float* xs = (float*)carve(N_NODES * 3 * 4);
  float* xt = (float*)carve(N_NODES * 3 * 4);
  float* hA = (float*)carve(N_NODES * 64 * 4);
  float* hB = (float*)carve(N_NODES * 64 * 4);
  float* agg = (float*)carve(N_NODES * 64 * 4);
  float* cacc = (float*)carve(N_NODES * 3 * 4);
  float* hs = (float*)carve(N_NODES * 35 * 4);
  float* ht = (float*)carve(N_NODES * 35 * 4);
  _Float16* At = (_Float16*)carve(48 * 8192 * 2);
  float* T1 = (float*)carve(48 * 2048 * 4);
  float* O0 = (float*)carve(48 * 128 * 4);
  float* O1 = (float*)carve(48 * 128 * 4);
  float* cs0 = (float*)carve(128 * 48 * 4);
  float* cs1 = (float*)carve(128 * 48 * 4);
  float* simb = (float*)carve(128 * 128 * 4);
  float* corr_acc = (float*)carve(256);

  auto run_graph = [&](const float* hin, const float* xin, const int* edges,
                       const float* eattr, float* xwork, float* hsout) {
    hipMemcpyAsync(xwork, xin, N_NODES * 3 * 4, hipMemcpyDeviceToDevice, stream);
    k_emb_in<<<N_NODES / 64, 128, 0, stream>>>(hin, emb_in_w, emb_in_b, hA);
    float* cur = hA; float* nxt = hB;
    for (int l = 0; l < 5; ++l) {
      hipMemsetAsync(agg, 0, N_NODES * 64 * 4, stream);
      hipMemsetAsync(cacc, 0, N_NODES * 3 * 4, stream);
      EdgeArgs ea;
      ea.edges = edges; ea.eattr = eattr; ea.h = cur; ea.x = xwork;
      ea.agg = agg; ea.cacc = cacc;
      ea.ew1 = L[l].ew1; ea.eb1 = L[l].eb1; ea.ew2 = L[l].ew2; ea.eb2 = L[l].eb2;
      ea.lng = L[l].lng; ea.lnb = L[l].lnb;
      ea.cw1 = L[l].cw1; ea.cb1 = L[l].cb1; ea.cw2 = L[l].cw2;
      k_edge<<<N_EDGES / 64, 128, 0, stream>>>(ea);
      k_node<<<N_NODES / 64, 128, 0, stream>>>(cur, agg, cacc,
          L[l].nw1, L[l].nb1, L[l].nw2, L[l].nb2, nxt, xwork);
      float* t = cur; cur = nxt; nxt = t;
    }
    k_emb_out<<<N_NODES / 64, 128, 0, stream>>>(cur, emb_out_w, emb_out_b, xwork, hsout);
  };

  run_graph(h_src, x_src, e_src, a_src, xs, hs);
  run_graph(h_tgt, x_tgt, e_tgt, a_tgt, xt, ht);

  hipMemsetAsync(corr_acc, 0, 4, stream);
  k_corr<<<4, 256, 0, stream>>>(hs, ht, corr, labels, corr_acc);

  for (int g = 0; g < 2; ++g) {
    const float* hsg = g ? ht : hs;
    float* Og = g ? O1 : O0;
    float* csg = g ? cs1 : cs0;
    k_transpose<<<(48 * 8192) / 256, 256, 0, stream>>>(hsg, At);
    k_comp1<<<dim3(32, 3), 128, 0, stream>>>(At, comp_w1, comp_b1, T1);
    k_comp2<<<dim3(2, 3), 128, 0, stream>>>(T1, comp_w2, comp_b2, Og);
    k_norm<<<1, 128, 0, stream>>>(Og, csg);
  }
  k_sim<<<64, 256, 0, stream>>>(cs0, cs1, simb);
  k_jacobi<<<1, 256, 0, stream>>>(simb, corr_acc, (float*)d_out);
}
